// TarsCoreBlock_55027120996831
// MI455X (gfx1250) — compile-verified, matrix operand order checked by reading, not run on hardware
//
#include <hip/hip_runtime.h>

// ---------------------------------------------------------------- constants
constexpr int BATCH    = 2;
constexpr int SEQLEN   = 4096;
constexpr int D_MODEL  = 768;
constexpr int D_STATE  = 64;
constexpr int D_CONV   = 4;
constexpr int HEADDIM  = 64;
constexpr int D_INNER  = 1536;
constexpr int NHEADS   = 24;               // D_INNER / HEADDIM
constexpr int CHUNK    = 64;
constexpr int CONV_DIM = 1664;             // D_INNER + 2*D_STATE
constexpr int D_MAMBA  = 3224;             // 2*D_INNER + 2*D_STATE + NHEADS
constexpr int D_RWKV   = 320;
constexpr int NPROJ    = D_MAMBA + D_RWKV; // 3544
constexpr int NTOK     = BATCH * SEQLEN;   // 8192
constexpr int NCHUNKS  = SEQLEN / CHUNK;   // 64

typedef __attribute__((ext_vector_type(16))) _Float16 v16h;
typedef __attribute__((ext_vector_type(8)))  float    v8f;
typedef __attribute__((ext_vector_type(8)))  _Float16 h8;   // 16-byte f16 chunk

// ---------------------------------------------------------------- helpers
__device__ __forceinline__ v8f wmma_f16(v16h a, v16h b, v8f c) {
  // D = A(16x32 f16) * B(32x16 f16) + C(16x16 f32)
  return __builtin_amdgcn_wmma_f32_16x16x32_f16(
      false, a, false, b, (short)0, c, false, false);
}

__device__ __forceinline__ float sigmoidf_(float x) { return 1.0f / (1.0f + __expf(-x)); }
__device__ __forceinline__ float siluf_(float x)    { return x * sigmoidf_(x); }
__device__ __forceinline__ float softplusf_(float x){ return (x > 20.f) ? x : log1pf(__expf(x)); }

// Fragment loader for A-operands (rows=M, cols=K) and for B-operands whose
// KxN matrix is stored transposed in LDS as [N][K].  16-bit A layout:
// lanes 0-15: M=lane, K groups {0..7,16..23}; lanes 16-31: M=lane-16, {8..15,24..31}.
// Per lane this is two contiguous 16B reads -> compiler emits 2x ds_load_b128.
__device__ __forceinline__ v16h ld_frag(const _Float16* base, int ld, int row0) {
  const int m    = threadIdx.x & 15;
  const int half = (threadIdx.x >> 4) & 1;
  const _Float16* p = base + (size_t)(row0 + m) * ld;
  v16h f;
#pragma unroll
  for (int j = 0; j < 8; ++j) {
    const int kk = (j < 4) ? (8 * half + 2 * j) : (16 + 8 * half + 2 * (j - 4));
    f[2 * j]     = p[kk];
    f[2 * j + 1] = p[kk + 1];
  }
  return f;
}

// Same, but scale each element by scale[k] (decay-weighted operand).
__device__ __forceinline__ v16h ld_frag_scaled(const _Float16* base, int ld, int row0,
                                               int kb, const float* scale) {
  const int m    = threadIdx.x & 15;
  const int half = (threadIdx.x >> 4) & 1;
  const _Float16* p = base + (size_t)(row0 + m) * ld + kb;
  v16h f;
#pragma unroll
  for (int j = 0; j < 8; ++j) {
    const int kk = (j < 4) ? (8 * half + 2 * j) : (16 + 8 * half + 2 * (j - 4));
    f[2 * j]     = (_Float16)((float)p[kk]     * scale[kb + kk]);
    f[2 * j + 1] = (_Float16)((float)p[kk + 1] * scale[kb + kk + 1]);
  }
  return f;
}

// ---------------------------------------------------------------- elementwise
__global__ __launch_bounds__(256) void k_cvt(const float* __restrict__ s,
                                             _Float16* __restrict__ d, int n) {
  int i = blockIdx.x * 256 + threadIdx.x;
  if (i < n) d[i] = (_Float16)s[i];
}

// u_mixed = u*tm + shift(u)*(1-tm)  -> f16
__global__ __launch_bounds__(256) void k_mix(const float* __restrict__ u,
                                             const float* __restrict__ tm,
                                             _Float16* __restrict__ out) {
  int i = blockIdx.x * 256 + threadIdx.x;
  if (i >= NTOK * D_MODEL) return;
  int row = i / D_MODEL, c = i - row * D_MODEL;
  int l = row & (SEQLEN - 1);
  float cur = u[i];
  float prv = (l > 0) ? u[i - D_MODEL] : 0.f;
  float t = tm[c];
  out[i] = (_Float16)(cur * t + prv * (1.f - t));
}

// dt = softplus(proj[:,3200:3224] + dt_bias)
__global__ __launch_bounds__(256) void k_dt(const float* __restrict__ proj,
                                            const float* __restrict__ dt_bias,
                                            float* __restrict__ dt) {
  int i = blockIdx.x * 256 + threadIdx.x;
  if (i >= NTOK * NHEADS) return;
  int row = i / NHEADS, hh = i - row * NHEADS;
  dt[i] = softplusf_(proj[(size_t)row * NPROJ + (D_INNER + CONV_DIM) + hh] + dt_bias[hh]);
}

// in-place: r -> sigmoid(r), w -> exp(-exp(w))
__global__ __launch_bounds__(256) void k_rwkv_prep(float* __restrict__ proj) {
  int i = blockIdx.x * 256 + threadIdx.x;
  if (i >= NTOK * 128) return;
  int row = i >> 7, c = i & 127;
  size_t idx = (size_t)row * NPROJ + D_MAMBA + (c < 64 ? c : 192 + (c - 64));
  float v = proj[idx];
  proj[idx] = (c < 64) ? sigmoidf_(v) : __expf(-__expf(v));
}

// depthwise causal conv (K=4) + silu over proj cols [1536,3200) -> xBC
__global__ __launch_bounds__(256) void k_conv(const float* __restrict__ proj,
                                              const float* __restrict__ cw,
                                              const float* __restrict__ cb,
                                              float* __restrict__ xBC) {
  int i = blockIdx.x * 256 + threadIdx.x;
  if (i >= NTOK * CONV_DIM) return;
  int row = i / CONV_DIM, ch = i - row * CONV_DIM;
  int l = row & (SEQLEN - 1);
  float acc = cb[ch];
#pragma unroll
  for (int t = 0; t < D_CONV; ++t) {
    int lp = l - (D_CONV - 1) + t;
    if (lp >= 0)
      acc += proj[(size_t)(row - (l - lp)) * NPROJ + D_INNER + ch] * cw[ch * D_CONV + t];
  }
  xBC[i] = siluf_(acc);
}

// concat [y_mamba | y_rwkv] -> f16
__global__ __launch_bounds__(256) void k_cat(const float* __restrict__ ym,
                                             const float* __restrict__ yr,
                                             _Float16* __restrict__ cat) {
  int i = blockIdx.x * 256 + threadIdx.x;
  if (i >= NTOK * 2 * D_INNER) return;
  int row = i / (2 * D_INNER), c = i - row * (2 * D_INNER);
  float v = (c < D_INNER) ? ym[(size_t)row * D_INNER + c]
                          : yr[(size_t)row * D_INNER + (c - D_INNER)];
  cat[i] = (_Float16)v;
}

// h = silu(t1 + b1) -> f16
__global__ __launch_bounds__(256) void k_h(const float* __restrict__ t1,
                                           const float* __restrict__ b1,
                                           _Float16* __restrict__ h) {
  int i = blockIdx.x * 256 + threadIdx.x;
  if (i >= NTOK * D_INNER) return;
  int c = i % D_INNER;
  h[i] = (_Float16)siluf_(t1[i] + b1[c]);
}

// gate/combine/silu(z)/layernorm -> f16 (one block per token)
__global__ __launch_bounds__(256) void k_final(const float* __restrict__ t2,
                                               const float* __restrict__ b2,
                                               const float* __restrict__ ym,
                                               const float* __restrict__ yr,
                                               const float* __restrict__ proj,
                                               const float* __restrict__ ln_g,
                                               const float* __restrict__ ln_b,
                                               _Float16* __restrict__ yln) {
  const int row = blockIdx.x;
  const int tid = threadIdx.x;
  __shared__ float red[256];
  __shared__ float s_mean, s_rstd;
  float vals[6];
  float s = 0.f;
#pragma unroll
  for (int i = 0; i < 6; ++i) {
    int c = i * 256 + tid;
    size_t idx = (size_t)row * D_INNER + c;
    float g = sigmoidf_(t2[idx] + b2[c]);
    float y = g * ym[idx] + (1.f - g) * yr[idx];
    float z = proj[(size_t)row * NPROJ + c];
    y *= siluf_(z);
    vals[i] = y;
    s += y;
  }
  red[tid] = s; __syncthreads();
  for (int o = 128; o > 0; o >>= 1) { if (tid < o) red[tid] += red[tid + o]; __syncthreads(); }
  if (tid == 0) s_mean = red[0] / (float)D_INNER;
  __syncthreads();
  float mean = s_mean, s2 = 0.f;
#pragma unroll
  for (int i = 0; i < 6; ++i) { float d = vals[i] - mean; s2 += d * d; }
  red[tid] = s2; __syncthreads();
  for (int o = 128; o > 0; o >>= 1) { if (tid < o) red[tid] += red[tid + o]; __syncthreads(); }
  if (tid == 0) s_rstd = rsqrtf(red[0] / (float)D_INNER + 1e-5f);
  __syncthreads();
  float rstd = s_rstd;
#pragma unroll
  for (int i = 0; i < 6; ++i) {
    int c = i * 256 + tid;
    yln[(size_t)row * D_INNER + c] =
        (_Float16)((vals[i] - mean) * rstd * ln_g[c] + ln_b[c]);
  }
}

// ---------------------------------------------------------------- GEMM
// C(MxN fp32) = A(MxK f16, row major) @ W(NxK f16, row major)^T
// 128x128x32 tiles, 8 waves (4M x 2N), each wave 2x4 WMMA tiles.
// M must be a multiple of 128 (true for all call sites); N guarded branchless;
// K % 32 == 0.  Ping-pong LDS (1 barrier / K-step) + register prefetch +
// distance-2 global_prefetch_b8.
__global__ __launch_bounds__(256) void k_gemm(const _Float16* __restrict__ A,
                                              const _Float16* __restrict__ W,
                                              float* __restrict__ C,
                                              int M, int N, int K) {
  constexpr int LDT = 40; // 32 + 8 pad (halves); 80B row stride, 16B aligned
  __shared__ _Float16 sA[2][128][LDT];
  __shared__ _Float16 sW[2][128][LDT];

  const int bn0 = blockIdx.x * 128;
  const int bm0 = blockIdx.y * 128;
  const int tid = threadIdx.x;
  const int wave = tid >> 5;
  const int wm = wave >> 1;  // 0..3 -> M offset *32
  const int wn = wave & 1;   // 0..1 -> N offset *64
  const int lrow = tid >> 2; // 0..63
  const int lseg = (tid & 3) * 8;

  // per-thread global source rows (fixed across K loop); W rows clamped so the
  // load is always in-bounds, then zero-selected (branchless, no exec masking)
  const int gm0 = bm0 + lrow, gm1 = bm0 + lrow + 64;
  const int gn0 = bn0 + lrow, gn1 = bn0 + lrow + 64;
  const bool n0ok = gn0 < N, n1ok = gn1 < N;
  const _Float16* pA0 = A + (size_t)gm0 * K + lseg;
  const _Float16* pA1 = A + (size_t)gm1 * K + lseg;
  const _Float16* pW0 = W + (size_t)(n0ok ? gn0 : 0) * K + lseg;
  const _Float16* pW1 = W + (size_t)(n1ok ? gn1 : 0) * K + lseg;

  v8f acc[2][4];
  v8f zero = {};
#pragma unroll
  for (int i = 0; i < 2; ++i)
#pragma unroll
    for (int j = 0; j < 4; ++j) acc[i][j] = zero;

  const h8 hz = {};
  h8 av0, av1, wv0, wv1;
  auto loadregs = [&](int k0) {
    av0 = *(const h8*)(pA0 + k0);
    av1 = *(const h8*)(pA1 + k0);
    h8 w0 = *(const h8*)(pW0 + k0);
    h8 w1 = *(const h8*)(pW1 + k0);
    wv0 = n0ok ? w0 : hz;   // v_cndmask, no branch
    wv1 = n1ok ? w1 : hz;
  };
  auto storelds = [&](int p) {
    *(h8*)&sA[p][lrow][lseg]      = av0;
    *(h8*)&sA[p][lrow + 64][lseg] = av1;
    *(h8*)&sW[p][lrow][lseg]      = wv0;
    *(h8*)&sW[p][lrow + 64][lseg] = wv1;
  };

  loadregs(0);
  storelds(0);
  __syncthreads();

  int p = 0;
  for (int k0 = 0; k0 < K; k0 += 32) {
    const bool more = (k0 + 32) < K;
    if (more) loadregs(k0 + 32);
    // distance-2 speculative prefetch into GL2 (dropped silently if OOB)
    __builtin_prefetch(pA0 + k0 + 64, 0, 0);
    __builtin_prefetch(pA1 + k0 + 64, 0, 0);
    __builtin_prefetch(pW0 + k0 + 64, 0, 0);
    __builtin_prefetch(pW1 + k0 + 64, 0, 0);

    v16h af[2], bf[4];
#pragma unroll
    for (int i = 0; i < 2; ++i) af[i] = ld_frag(&sA[p][0][0], LDT, wm * 32 + i * 16);
#pragma unroll
    for (int j = 0; j < 4; ++j) bf[j] = ld_frag(&sW[p][0][0], LDT, wn * 64 + j * 16);
#pragma unroll
    for (int i = 0; i < 2; ++i)
#pragma unroll
      for (int j = 0; j < 4; ++j) acc[i][j] = wmma_f16(af[i], bf[j], acc[i][j]);

    if (more) {
      storelds(p ^ 1);
      __syncthreads();
      p ^= 1;
    }
  }

  const int lane = tid & 31;
  const int half = lane >> 4;
  const int n15 = lane & 15;
#pragma unroll
  for (int i = 0; i < 2; ++i)
#pragma unroll
    for (int j = 0; j < 4; ++j)
#pragma unroll
      for (int r = 0; r < 8; ++r) {
        int gRow = bm0 + wm * 32 + i * 16 + r + 8 * half;
        int gCol = bn0 + wn * 64 + j * 16 + n15;
        if (gCol < N) C[(size_t)gRow * N + gCol] = acc[i][j][r];
      }
}

// ---------------------------------------------------------------- SSD scan
// One workgroup per (batch, head); sequential over 64 chunks; all four 64^3
// matmuls per chunk with v_wmma_f32_16x16x32_f16; fp32 carried state in LDS.
__global__ __launch_bounds__(256) void k_ssd(const float* __restrict__ xBC,
                                             const float* __restrict__ dt,
                                             const float* __restrict__ A_log,
                                             const float* __restrict__ D_skip,
                                             float* __restrict__ Y) {
  constexpr int LH = 72; // padded f16 row stride (144B, 16B aligned)
  __shared__ _Float16 sC [64][LH];  // C[l][n]
  __shared__ _Float16 sB [64][LH];  // B[l][n]   (B-op storage [N=s][K=n] for G)
  __shared__ _Float16 sBT[64][LH];  // B^T[n][l] (B-op storage [N=n][K=l] for state)
  __shared__ _Float16 sXT[64][LH];  // (x*dt)^T [p][l]
  __shared__ _Float16 sM [64][LH];  // masked decay * G  [l][s]
  __shared__ _Float16 sS16[64][LH]; // carried state^T f16 [p][n]
  __shared__ float    sSf[64][68];  // carried state^T f32 [p][n]
  __shared__ float    sAc[64], sDecL[64], sEAc[64], sDt[64];

  const int b = blockIdx.x / NHEADS;
  const int h = blockIdx.x - b * NHEADS;
  const int tid = threadIdx.x;
  const int wave = tid >> 5;
  const int lane = tid & 31;
  const int half = lane >> 4;
  const int n15 = lane & 15;
  const int lrow = tid >> 2;          // 0..63
  const int cseg = (tid & 3) * 16;    // 0,16,32,48

  const float Ah = -__expf(A_log[h]);
  const float Dh = D_skip[h];

  for (int i = tid; i < 64 * 64; i += 256) sSf[i >> 6][i & 63] = 0.f;

  for (int c = 0; c < NCHUNKS; ++c) {
    const int rowbase = b * SEQLEN + c * CHUNK;
    // ---- phase 1: vectorized global loads, row-major LDS tiles as b128 stores
    float xv[16], bv[16], cv[16];
    {
      const float* rowp = xBC + (size_t)(rowbase + lrow) * CONV_DIM;
      __builtin_prefetch(rowp + CHUNK * CONV_DIM + h * HEADDIM + cseg, 0, 0);
#pragma unroll
      for (int i4 = 0; i4 < 4; ++i4) {
        float4 a  = *(const float4*)(rowp + h * HEADDIM + cseg + i4 * 4);
        float4 b4 = *(const float4*)(rowp + D_INNER + cseg + i4 * 4);
        float4 c4 = *(const float4*)(rowp + D_INNER + D_STATE + cseg + i4 * 4);
        xv[i4*4+0]=a.x;  xv[i4*4+1]=a.y;  xv[i4*4+2]=a.z;  xv[i4*4+3]=a.w;
        bv[i4*4+0]=b4.x; bv[i4*4+1]=b4.y; bv[i4*4+2]=b4.z; bv[i4*4+3]=b4.w;
        cv[i4*4+0]=c4.x; cv[i4*4+1]=c4.y; cv[i4*4+2]=c4.z; cv[i4*4+3]=c4.w;
      }
      h8 t0, t1;
#pragma unroll
      for (int i = 0; i < 8; ++i) { t0[i] = (_Float16)bv[i]; t1[i] = (_Float16)bv[8+i]; }
      *(h8*)&sB[lrow][cseg] = t0;  *(h8*)&sB[lrow][cseg + 8] = t1;
#pragma unroll
      for (int i = 0; i < 8; ++i) { t0[i] = (_Float16)cv[i]; t1[i] = (_Float16)cv[8+i]; }
      *(h8*)&sC[lrow][cseg] = t0;  *(h8*)&sC[lrow][cseg + 8] = t1;
#pragma unroll
      for (int i = 0; i < 16; ++i) sBT[cseg + i][lrow] = (_Float16)bv[i];
      if (tid < 64) sDt[tid] = dt[(size_t)(rowbase + tid) * NHEADS + h];
    }
    __syncthreads();
    // ---- phase 3: cumsum (thread 0), X^T scaled by dt, state f16 snapshot
    if (tid == 0) {
      float acc = 0.f;
      for (int l = 0; l < 64; ++l) { acc += Ah * sDt[l]; sAc[l] = acc; }
    }
    {
      float dts = sDt[lrow];
#pragma unroll
      for (int i = 0; i < 16; ++i) sXT[cseg + i][lrow] = (_Float16)(xv[i] * dts);
      h8 t0, t1;
#pragma unroll
      for (int i = 0; i < 8; ++i) {
        t0[i] = (_Float16)sSf[lrow][cseg + i];
        t1[i] = (_Float16)sSf[lrow][cseg + 8 + i];
      }
      *(h8*)&sS16[lrow][cseg] = t0;  *(h8*)&sS16[lrow][cseg + 8] = t1;
    }
    __syncthreads();
    if (tid < 64) {
      float last = sAc[63];
      sDecL[tid] = __expf(last - sAc[tid]);
      sEAc[tid]  = __expf(sAc[tid]);
    }
    __syncthreads();

    // tiles: wave handles tile ids 2w, 2w+1 of a 4x4 tile grid
    // ---- G = C @ B^T, then mask*decay -> sM
#pragma unroll
    for (int q = 0; q < 2; ++q) {
      int t = wave * 2 + q, mt = t >> 2, nt = t & 3;
      v8f g = {};
#pragma unroll
      for (int kb = 0; kb < 64; kb += 32) {
        v16h a  = ld_frag(&sC[0][0] + kb, LH, mt * 16);
        v16h bb = ld_frag(&sB[0][0] + kb, LH, nt * 16);
        g = wmma_f16(a, bb, g);
      }
#pragma unroll
      for (int r = 0; r < 8; ++r) {
        int l = mt * 16 + r + 8 * half;
        int s = nt * 16 + n15;
        float v = (s <= l) ? g[r] * __expf(sAc[l] - sAc[s]) : 0.f;
        sM[l][s] = (_Float16)v;
      }
    }
    __syncthreads();

    // ---- Y = exp(Ac[l])*(C @ S) + M @ X ; St = (X^T*decL) @ B
#pragma unroll
    for (int q = 0; q < 2; ++q) {
      int t = wave * 2 + q, mt = t >> 2, nt = t & 3;
      v8f y = {};
#pragma unroll
      for (int kb = 0; kb < 64; kb += 32) {
        v16h a  = ld_frag(&sC[0][0] + kb, LH, mt * 16);
        v16h bb = ld_frag(&sS16[0][0] + kb, LH, nt * 16);
        y = wmma_f16(a, bb, y);
      }
#pragma unroll
      for (int r = 0; r < 8; ++r) y[r] *= sEAc[mt * 16 + r + 8 * half];
#pragma unroll
      for (int kb = 0; kb < 64; kb += 32) {
        v16h a  = ld_frag(&sM[0][0] + kb, LH, mt * 16);
        v16h bb = ld_frag(&sXT[0][0] + kb, LH, nt * 16);
        y = wmma_f16(a, bb, y);
      }
#pragma unroll
      for (int r = 0; r < 8; ++r) {
        int l = mt * 16 + r + 8 * half;
        int p = nt * 16 + n15;
        size_t row = (size_t)(rowbase + l);
        float xvg = xBC[row * CONV_DIM + h * HEADDIM + p];
        Y[row * D_INNER + h * HEADDIM + p] = y[r] + Dh * xvg;
      }
      v8f st = {};
#pragma unroll
      for (int kb = 0; kb < 64; kb += 32) {
        v16h a  = ld_frag_scaled(&sXT[0][0], LH, mt * 16, kb, sDecL);
        v16h bb = ld_frag(&sBT[0][0] + kb, LH, nt * 16);
        st = wmma_f16(a, bb, st);
      }
      float eLast = __expf(sAc[63]);
#pragma unroll
      for (int r = 0; r < 8; ++r) {
        int p = mt * 16 + r + 8 * half;
        int n = nt * 16 + n15;
        sSf[p][n] = sSf[p][n] * eLast + st[r];
      }
    }
    __syncthreads();
  }
}

// ---------------------------------------------------------------- WKV scan
// grid = BATCH*32 blocks of 1 wave; each wave owns 2 state rows, columns
// split 2 per lane; serial over L with wave-local reductions only.
// Latency-bound -> prefetch t+8's row to keep L0 warm.
__global__ __launch_bounds__(32) void k_wkv(const float* __restrict__ proj,
                                            float* __restrict__ y) {
  const int b = blockIdx.x >> 5;
  const int w = blockIdx.x & 5 * 6 + 1; // placeholder (overwritten below)
  (void)w;
  const int wv = blockIdx.x & 31;
  const int lane = threadIdx.x;
  const int i0 = 2 * wv, i1 = 2 * wv + 1;
  float s00 = 0.f, s01 = 0.f, s10 = 0.f, s11 = 0.f;
  for (int t = 0; t < SEQLEN; ++t) {
    const float* base = proj + (size_t)(b * SEQLEN + t) * NPROJ + D_MAMBA;
    __builtin_prefetch(base + 8 * NPROJ + lane * 8, 0, 0);
    float r0 = base[i0],        r1 = base[i1];
    float k0 = base[64 + i0],   k1 = base[64 + i1];
    float w0 = base[192 + i0],  w1 = base[192 + i1];
    float bb0 = base[256 + i0], bb1 = base[256 + i1];
    float vA = base[128 + lane], vB = base[128 + 32 + lane];
    float kA = base[64 + lane],  kB = base[64 + 32 + lane];
    float kv0 = k0 * bb0, kv1 = k1 * bb1;
    s00 = s00 * w0 + kv0 * vA;  s01 = s01 * w0 + kv0 * vB;
    s10 = s10 * w1 + kv1 * vA;  s11 = s11 * w1 + kv1 * vB;
    float d0 = s00 * kA + s01 * kB;
    float d1 = s10 * kA + s11 * kB;
#pragma unroll
    for (int off = 16; off > 0; off >>= 1) {
      d0 += __shfl_xor(d0, off);
      d1 += __shfl_xor(d1, off);
    }
    if (lane == 0) {
      float* yo = y + (size_t)(b * SEQLEN + t) * D_STATE;
      yo[i0] = r0 * d0;
      yo[i1] = r1 * d1;
    }
  }
}

// ---------------------------------------------------------------- launcher
extern "C" void kernel_launch(void* const* d_in, const int* in_sizes, int n_in,
                              void* d_out, int out_size, void* d_ws, size_t ws_size,
                              hipStream_t stream) {
  const float* u         = (const float*)d_in[0];
  const float* in_proj_w = (const float*)d_in[1];
  const float* conv_w    = (const float*)d_in[2];
  const float* conv_b    = (const float*)d_in[3];
  const float* dt_bias   = (const float*)d_in[4];
  const float* A_log     = (const float*)d_in[5];
  const float* D_skip    = (const float*)d_in[6];
  const float* time_mix  = (const float*)d_in[7];
  const float* wkv_up_w  = (const float*)d_in[8];
  const float* fg_w1     = (const float*)d_in[9];
  const float* fg_b1     = (const float*)d_in[10];
  const float* fg_w2     = (const float*)d_in[11];
  const float* fg_b2     = (const float*)d_in[12];
  const float* ln_g      = (const float*)d_in[13];
  const float* ln_b      = (const float*)d_in[14];
  const float* out_proj_w= (const float*)d_in[15];
  float* out = (float*)d_out;

  char* ws = (char*)d_ws;
  size_t off = 0;
  auto alloc = [&](size_t bytes) -> void* {
    void* p = ws + off;
    off = (off + bytes + 255) & ~(size_t)255;
    return p;
  };

  _Float16* Win16   = (_Float16*)alloc((size_t)NPROJ * D_MODEL * 2);
  _Float16* Wfg1_16 = (_Float16*)alloc((size_t)D_INNER * 2 * D_INNER * 2);
  _Float16* Wfg2_16 = (_Float16*)alloc((size_t)D_INNER * D_INNER * 2);
  _Float16* Wwkv16  = (_Float16*)alloc((size_t)D_INNER * D_STATE * 2);
  _Float16* Wout16  = (_Float16*)alloc((size_t)D_MODEL * D_INNER * 2);
  _Float16* umix16  = (_Float16*)alloc((size_t)NTOK * D_MODEL * 2);
  float*    proj    = (float*)   alloc((size_t)NTOK * NPROJ * 4);
  float*    dt      = (float*)   alloc((size_t)NTOK * NHEADS * 4);
  float*    xBC     = (float*)   alloc((size_t)NTOK * CONV_DIM * 4);
  float*    y_mamba = (float*)   alloc((size_t)NTOK * D_INNER * 4);
  float*    wkv32   = (float*)   alloc((size_t)NTOK * D_STATE * 4);
  _Float16* wkv16   = (_Float16*)alloc((size_t)NTOK * D_STATE * 2);
  float*    y_rwkv  = (float*)   alloc((size_t)NTOK * D_INNER * 4);
  _Float16* cat16   = (_Float16*)alloc((size_t)NTOK * 2 * D_INNER * 2);
  float*    t1      = (float*)   alloc((size_t)NTOK * D_INNER * 4);
  _Float16* h16     = (_Float16*)alloc((size_t)NTOK * D_INNER * 2);
  _Float16* yln16   = (_Float16*)alloc((size_t)NTOK * D_INNER * 2);
  float*    t2      = t1; // reuse: t1 consumed before fg2 output

  auto cvt = [&](const float* s, _Float16* d, int n) {
    k_cvt<<<dim3((n + 255) / 256), dim3(256), 0, stream>>>(s, d, n);
  };
  cvt(in_proj_w, Win16,   NPROJ * D_MODEL);
  cvt(fg_w1,     Wfg1_16, D_INNER * 2 * D_INNER);
  cvt(fg_w2,     Wfg2_16, D_INNER * D_INNER);
  cvt(wkv_up_w,  Wwkv16,  D_INNER * D_STATE);
  cvt(out_proj_w,Wout16,  D_MODEL * D_INNER);

  k_mix<<<dim3((NTOK * D_MODEL + 255) / 256), dim3(256), 0, stream>>>(u, time_mix, umix16);

  auto gemm = [&](const _Float16* A, const _Float16* W, float* C, int M, int N, int K) {
    dim3 g((N + 127) / 128, M / 128);
    k_gemm<<<g, dim3(256), 0, stream>>>(A, W, C, M, N, K);
  };

  gemm(umix16, Win16, proj, NTOK, NPROJ, D_MODEL);

  k_dt<<<dim3((NTOK * NHEADS + 255) / 256), dim3(256), 0, stream>>>(proj, dt_bias, dt);
  k_rwkv_prep<<<dim3((NTOK * 128 + 255) / 256), dim3(256), 0, stream>>>(proj);
  k_conv<<<dim3((NTOK * CONV_DIM + 255) / 256), dim3(256), 0, stream>>>(proj, conv_w, conv_b, xBC);

  k_ssd<<<dim3(BATCH * NHEADS), dim3(256), 0, stream>>>(xBC, dt, A_log, D_skip, y_mamba);

  k_wkv<<<dim3(BATCH * 32), dim3(32), 0, stream>>>(proj, wkv32);
  cvt(wkv32, wkv16, NTOK * D_STATE);
  gemm(wkv16, Wwkv16, y_rwkv, NTOK, D_INNER, D_STATE);

  k_cat<<<dim3((NTOK * 2 * D_INNER + 255) / 256), dim3(256), 0, stream>>>(y_mamba, y_rwkv, cat16);
  gemm(cat16, Wfg1_16, t1, NTOK, D_INNER, 2 * D_INNER);
  k_h<<<dim3((NTOK * D_INNER + 255) / 256), dim3(256), 0, stream>>>(t1, fg_b1, h16);
  gemm(h16, Wfg2_16, t2, NTOK, D_INNER, D_INNER);

  k_final<<<dim3(NTOK), dim3(256), 0, stream>>>(t2, fg_b2, y_mamba, y_rwkv, proj,
                                                ln_g, ln_b, yln16);
  gemm(yln16, Wout16, out, NTOK, D_MODEL, D_INNER);
  (void)in_sizes; (void)n_in; (void)out_size; (void)ws_size;
}
